// ConvFastCircular_77137612636252
// MI455X (gfx1250) — matrix operand order repaired and checked by reading, not compiled, not run on hardware
//
#include <hip/hip_runtime.h>

#define CH   64
#define FOUT 256
#define KTILES 18            // 576 / 32
#define NTILES 16            // 256 / 16
#define NPIX   (16 * 128 * 128)   // B*H*W = 262144 output rows

typedef __attribute__((ext_vector_type(16))) __bf16 v16bf;
typedef __attribute__((ext_vector_type(8)))  __bf16 v8bf;
typedef __attribute__((ext_vector_type(8)))  float  v8f;

// ---------------------------------------------------------------------------
// Pack kernel [576, 256] fp32 -> bf16 fragments in exact WMMA B-operand lane
// order.  Fragment (t, n): 32 lanes x 16 bf16 (32 B / lane, contiguous).
//   lane < 16 : col = n*16 + lane,      elements j = B[t*32 + j + (j>=8 ? 8 : 0),     col]
//   lane >=16 : col = n*16 + lane - 16, elements j = B[t*32 + 8 + j + (j>=8 ? 8 : 0), col]
// ---------------------------------------------------------------------------
__global__ void pack_b_kernel(const float* __restrict__ kern,
                              unsigned short* __restrict__ bpack) {
  int tid = blockIdx.x * blockDim.x + threadIdx.x;   // one thread per (t,n,lane)
  if (tid >= KTILES * NTILES * 32) return;
  int lane = tid & 31;
  int n    = (tid >> 5) & 15;
  int t    = tid >> 9;
  int col  = n * 16 + (lane & 15);
  int kk   = t * 32 + ((lane >> 4) << 3);
  __bf16* dst = reinterpret_cast<__bf16*>(bpack) + (size_t)tid * 16;
#pragma unroll
  for (int j = 0; j < 16; ++j)
    dst[j] = (__bf16)kern[(size_t)(kk + j + (j >= 8 ? 8 : 0)) * FOUT + col];
}

// ---------------------------------------------------------------------------
// One-shot fp32 -> bf16 conversion of x (NHWC layout preserved).
// Each thread converts 8 elements: 2x b128 load, 4x cvt_pk, 1x b128 store.
// ---------------------------------------------------------------------------
__global__ void convert_x_kernel(const float* __restrict__ x,
                                 unsigned short* __restrict__ xbf) {
  size_t tid  = (size_t)blockIdx.x * blockDim.x + threadIdx.x;
  size_t base = tid * 8;
  if (base >= (size_t)NPIX * CH) return;
  const float4* q = reinterpret_cast<const float4*>(x + base);
  const float4 q0 = q[0], q1 = q[1];
  v8bf o;
  o[0] = (__bf16)q0.x; o[1] = (__bf16)q0.y; o[2] = (__bf16)q0.z; o[3] = (__bf16)q0.w;
  o[4] = (__bf16)q1.x; o[5] = (__bf16)q1.y; o[6] = (__bf16)q1.z; o[7] = (__bf16)q1.w;
  reinterpret_cast<v8bf*>(xbf)[tid] = o;
}

// ---------------------------------------------------------------------------
// GEMM: out[p, f] = sum_{s,c} x[b, (h-sh)%128, (w-sw)%128, c] * K[s*64+c, f]
// Block = 256 threads = 8 waves (4 M-waves x 2 N-waves), WG tile 128M x 256N.
// Wave tile: 32M x 128N -> 2x8 accumulators, 16 WMMAs / K-tile.
// PRECONV=true  : A loaded from pre-converted bf16 x (2x b128 per fragment)
// PRECONV=false : A loaded fp32 + converted on the fly (4x b128 + 8 cvt_pk)
// ---------------------------------------------------------------------------
template <bool PRECONV>
__global__ __launch_bounds__(256)
void conv_gemm_kernel(const float* __restrict__ x,
                      const unsigned short* __restrict__ xbf_raw,
                      const unsigned short* __restrict__ bpack_raw,
                      const float* __restrict__ bias,
                      float* __restrict__ out) {
  const int lane   = threadIdx.x & 31;
  const int waveId = threadIdx.x >> 5;
  const int waveM  = waveId >> 1;          // 0..3
  const int waveN  = waveId & 1;           // 0..1

  const int mBase  = blockIdx.x * 128 + waveM * 32;  // 32 rows / wave
  const int nBase  = waveN * 128;                    // 128 cols / wave
  const int khalf8 = (lane >> 4) << 3;               // 0 or 8
  const int rlane  = lane & 15;

  // Decompose the two M-tile row pixels this lane owns.
  int h0[2], w0[2], bofs[2];
#pragma unroll
  for (int mt = 0; mt < 2; ++mt) {
    int p    = mBase + mt * 16 + rlane;
    w0[mt]   = p & 127;
    h0[mt]   = (p >> 7) & 127;
    bofs[mt] = p & ~16383;                 // (b << 14)
  }

  const v16bf* __restrict__ bpack = reinterpret_cast<const v16bf*>(bpack_raw);
  const __bf16* __restrict__ xbf  = reinterpret_cast<const __bf16*>(xbf_raw);
  const int nTile0 = waveN * 8;            // first global N-tile (of 8)

  v8f acc[2][8];
#pragma unroll
  for (int mt = 0; mt < 2; ++mt)
#pragma unroll
    for (int nt = 0; nt < 8; ++nt)
      acc[mt][nt] = {};

  for (int t = 0; t < KTILES; ++t) {
    const int s   = t >> 1;                // shift index 0..8
    const int shh = (s % 3) - 1;           // meshgrid 'xy': fast axis -> H shift
    const int shw = (s / 3) - 1;           //                slow axis -> W shift
    const int c0  = ((t & 1) << 5) + khalf8;

    // ---- A fragments: contiguous 8+8 channels of one shifted pixel --------
    v16bf afrag[2];
#pragma unroll
    for (int mt = 0; mt < 2; ++mt) {
      const int hh   = (h0[mt] - shh) & 127;
      const int ww   = (w0[mt] - shw) & 127;
      const int srcp = bofs[mt] | (hh << 7) | ww;
      v16bf a;
      if (PRECONV) {
        const v8bf* q = reinterpret_cast<const v8bf*>(
            xbf + (size_t)srcp * CH + c0);
        const v8bf lo = q[0];              // K_rel = khalf8 + 0..7
        const v8bf hi = q[2];              // K_rel = khalf8 + 16..23
#pragma unroll
        for (int i = 0; i < 8; ++i) { a[i] = lo[i]; a[8 + i] = hi[i]; }
      } else {
        const float4* q = reinterpret_cast<const float4*>(
            x + (size_t)srcp * CH + c0);
        const float4 q0 = q[0], q1 = q[1];
        const float4 q2 = q[4], q3 = q[5];
        a[0]  = (__bf16)q0.x; a[1]  = (__bf16)q0.y;
        a[2]  = (__bf16)q0.z; a[3]  = (__bf16)q0.w;
        a[4]  = (__bf16)q1.x; a[5]  = (__bf16)q1.y;
        a[6]  = (__bf16)q1.z; a[7]  = (__bf16)q1.w;
        a[8]  = (__bf16)q2.x; a[9]  = (__bf16)q2.y;
        a[10] = (__bf16)q2.z; a[11] = (__bf16)q2.w;
        a[12] = (__bf16)q3.x; a[13] = (__bf16)q3.y;
        a[14] = (__bf16)q3.z; a[15] = (__bf16)q3.w;
      }
      afrag[mt] = a;
    }

    // Hint next K-tile's B block toward the caches.
    if (t + 1 < KTILES)
      __builtin_prefetch(&bpack[((size_t)(t + 1) * NTILES + nTile0) * 32 + lane], 0, 3);

    // ---- B fragments + WMMAs ---------------------------------------------
#pragma unroll
    for (int nt = 0; nt < 8; ++nt) {
      const v16bf bfrag = bpack[((size_t)t * NTILES + (nTile0 + nt)) * 32 + lane];
      acc[0][nt] = __builtin_amdgcn_wmma_f32_16x16x32_bf16(
          false, afrag[0], false, bfrag, (short)0, acc[0][nt], false, false);
      acc[1][nt] = __builtin_amdgcn_wmma_f32_16x16x32_bf16(
          false, afrag[1], false, bfrag, (short)0, acc[1][nt], false, false);
    }
  }

  // ---- Store: C/D layout = VGPR r -> row (r + 8*(lane>=16)), col = lane%16 -
  const int rowOff = khalf8;               // 0 or 8
#pragma unroll
  for (int nt = 0; nt < 8; ++nt) {
    const int col  = nBase + nt * 16 + rlane;
    const float bv = bias[col];
#pragma unroll
    for (int mt = 0; mt < 2; ++mt) {
      const int rbase = mBase + mt * 16 + rowOff;
#pragma unroll
      for (int r = 0; r < 8; ++r)
        out[(size_t)(rbase + r) * FOUT + col] = acc[mt][nt][r] + bv;
    }
  }
}

extern "C" void kernel_launch(void* const* d_in, const int* in_sizes, int n_in,
                              void* d_out, int out_size, void* d_ws, size_t ws_size,
                              hipStream_t stream) {
  (void)in_sizes; (void)n_in; (void)out_size;
  const float* x    = (const float*)d_in[0];
  const float* kern = (const float*)d_in[1];
  const float* bias = (const float*)d_in[2];
  float* out = (float*)d_out;

  const size_t bpackBytes = (size_t)KTILES * NTILES * 32 * 16 * 2;  // 288 KB
  const size_t xbfBytes   = (size_t)NPIX * CH * 2;                  // 33.5 MB
  unsigned short* bpack = (unsigned short*)d_ws;
  unsigned short* xbf   = (unsigned short*)((char*)d_ws + bpackBytes);

  // 1) pack weights into WMMA B-fragment order
  const int packThreads = KTILES * NTILES * 32;    // 9216
  pack_b_kernel<<<(packThreads + 255) / 256, 256, 0, stream>>>(kern, bpack);

  if (ws_size >= bpackBytes + xbfBytes) {
    // 2) one-shot x fp32->bf16 conversion (removes 18x redundant converts
    //    and halves A-side L2 traffic in the GEMM)
    const size_t cvtThreads = (size_t)NPIX * CH / 8;  // 2097152
    convert_x_kernel<<<(unsigned)(cvtThreads / 256), 256, 0, stream>>>(x, xbf);
    // 3) circular-conv GEMM, bf16 A path
    conv_gemm_kernel<true><<<NPIX / 128, 256, 0, stream>>>(x, xbf, bpack, bias, out);
  } else {
    // Fallback: convert A on the fly (workspace too small for bf16 copy of x)
    conv_gemm_kernel<false><<<NPIX / 128, 256, 0, stream>>>(x, nullptr, bpack, bias, out);
  }
}
